// RTFSBlock_2989297238425
// MI455X (gfx1250) — matrix operand order repaired and checked by reading, not compiled, not run on hardware
//
#include <hip/hip_runtime.h>

typedef __attribute__((ext_vector_type(16))) _Float16 v16h;
typedef __attribute__((ext_vector_type(8)))  _Float16 h8;
typedef __attribute__((ext_vector_type(4)))  _Float16 h4;
typedef __attribute__((ext_vector_type(8)))  float    v8f;

__device__ __forceinline__ float sigmoidf_(float x) { return 1.f / (1.f + __expf(-x)); }

// Build a 16xf16 WMMA fragment from two 16-byte LDS chunks (ds_load_b128 x2).
__device__ __forceinline__ v16h frag16(const _Float16* row, int c0, int c1)
{
  const h8* p = (const h8*)row;
  h8 x = p[c0], y = p[c1];
  return __builtin_shufflevector(x, y, 0, 1, 2, 3, 4, 5, 6, 7,
                                 8, 9, 10, 11, 12, 13, 14, 15);
}

// ======================================================================
// Generic strided / 2-level-batched GEMM:  C = alpha * A*B + bias
// A: (M,K) via (a_ms,a_ks); B: (K,N) via (b_ks,b_ns); C: (M,N) via (c_ms,c_ns)
// batch base = zo*{a,b,c}_bo + zi*{a,b,c}_bi, grid.z = nbo*nbi.
// FP16 inputs staged through LDS (double-buffered, K-step 64), FP32
// accumulation with v_wmma_f32_16x16x32_f16 (2 per K-step).
// Block = 128 threads = 4 waves, block tile 64x16, one 16x16 tile per wave.
// Stride-specialized float4 global loads; the GEMMs in this net, except
// attention scores/output, hit the unguarded fast paths (M%64==0, K%64==0, N%16==0).
// ======================================================================
__global__ __launch_bounds__(128) void k_gemm_wmma(
    const float* __restrict__ A, const float* __restrict__ Bm,
    float* __restrict__ C, const float* __restrict__ bias,
    int M, int N, int K, float alpha,
    long long a_ms, long long a_ks, long long a_bo, long long a_bi,
    long long b_ks, long long b_ns, long long b_bo, long long b_bi,
    long long c_ms, long long c_ns, long long c_bo, long long c_bi,
    int nbi)
{
  // 144-byte row pitch: 16B aligned for b128 LDS ops, staggers banks.
  __shared__ __align__(16) _Float16 As[2][64][72];
  __shared__ __align__(16) _Float16 Bs[2][16][72];   // transposed: [n][k]
  const int z  = blockIdx.z;
  const int zo = z / nbi, zi = z % nbi;
  const float* Ab = A  + (long long)zo * a_bo + (long long)zi * a_bi;
  const float* Bb = Bm + (long long)zo * b_bo + (long long)zi * b_bi;
  float*       Cb = C  + (long long)zo * c_bo + (long long)zi * c_bi;
  const int tid  = threadIdx.x;
  const int wave = tid >> 5;
  const int lane = tid & 31;
  const int lh   = (lane >> 4) & 1;   // lane half
  const int lr   = lane & 15;
  const int mblk = blockIdx.y * 64;
  const int n0   = blockIdx.x * 16;

  auto load_stage = [&](int k0, int st) {
    const bool fullK = (k0 + 64 <= K);
    const bool fullA = fullK && (mblk + 64 <= M);
    if (fullA && a_ks == 1 && ((a_ms & 3) == 0)) {
      // rows contiguous in K: float4 along K
#pragma unroll
      for (int q = 0; q < 8; ++q) {
        int u = q * 128 + tid;           // 64 rows x 16 k-quads
        int r = u >> 4, kq = u & 15;
        float4 v = *(const float4*)(Ab + (long long)(mblk + r) * a_ms + (k0 + kq * 4));
        h4 hv = { (_Float16)v.x, (_Float16)v.y, (_Float16)v.z, (_Float16)v.w };
        *(h4*)&As[st][r][kq * 4] = hv;
      }
    } else if (fullA && a_ms == 1 && ((a_ks & 3) == 0)) {
      // contiguous in M (NCHW conv1x1): float4 along M
#pragma unroll
      for (int q = 0; q < 8; ++q) {
        int u = q * 128 + tid;           // 16 row-quads x 64 k
        int mq = u & 15, kk = u >> 4;
        float4 v = *(const float4*)(Ab + (mblk + mq * 4) + (long long)(k0 + kk) * a_ks);
        As[st][mq * 4 + 0][kk] = (_Float16)v.x;
        As[st][mq * 4 + 1][kk] = (_Float16)v.y;
        As[st][mq * 4 + 2][kk] = (_Float16)v.z;
        As[st][mq * 4 + 3][kk] = (_Float16)v.w;
      }
    } else {
      for (int i = tid; i < 64 * 64; i += 128) {
        int r = i >> 6, kk = i & 63;
        int gm = mblk + r, gk = k0 + kk;
        float v = 0.f;
        if (gm < M && gk < K) v = Ab[(long long)gm * a_ms + (long long)gk * a_ks];
        As[st][r][kk] = (_Float16)v;
      }
    }
    const bool fullB = fullK && (n0 + 16 <= N);
    if (fullB && b_ks == 1 && ((b_ns & 3) == 0)) {
#pragma unroll
      for (int q = 0; q < 2; ++q) {
        int u = q * 128 + tid;           // 16 n x 16 k-quads
        int n = u >> 4, kq = u & 15;
        float4 v = *(const float4*)(Bb + (k0 + kq * 4) + (long long)(n0 + n) * b_ns);
        h4 hv = { (_Float16)v.x, (_Float16)v.y, (_Float16)v.z, (_Float16)v.w };
        *(h4*)&Bs[st][n][kq * 4] = hv;
      }
    } else if (fullB && b_ns == 1 && ((b_ks & 3) == 0)) {
#pragma unroll
      for (int q = 0; q < 2; ++q) {
        int u = q * 128 + tid;           // 64 k x 4 n-quads
        int kk = u >> 2, nq = u & 3;
        float4 v = *(const float4*)(Bb + (long long)(k0 + kk) * b_ks + (n0 + nq * 4));
        Bs[st][nq * 4 + 0][kk] = (_Float16)v.x;
        Bs[st][nq * 4 + 1][kk] = (_Float16)v.y;
        Bs[st][nq * 4 + 2][kk] = (_Float16)v.z;
        Bs[st][nq * 4 + 3][kk] = (_Float16)v.w;
      }
    } else {
      for (int i = tid; i < 16 * 64; i += 128) {
        int n = i >> 6, kk = i & 63;
        int gn = n0 + n, gk = k0 + kk;
        float v = 0.f;
        if (gn < N && gk < K) v = Bb[(long long)gk * b_ks + (long long)gn * b_ns];
        Bs[st][n][kk] = (_Float16)v;
      }
    }
  };

  v8f acc = {};
  load_stage(0, 0);
  __syncthreads();
  int st = 0;
  for (int k0 = 0; k0 < K; k0 += 64) {
    if (k0 + 64 < K) load_stage(k0 + 64, st ^ 1);
    const _Float16* arow = &As[st][wave * 16 + lr][0];
    const _Float16* brow = &Bs[st][lr][0];
    // A layout: lane<16 -> K {0..7,16..23}(+kk0), lane>=16 -> K {8..15,24..31}
    // B layout: lane<16 -> K {0..7,16..23}(+kk0), lane>=16 -> K {8..15,24..31}, n=lane&15
    v16h va0 = frag16(arow, lh, 2 + lh);
    v16h vb0 = frag16(brow, lh, 2 + lh);
    acc = __builtin_amdgcn_wmma_f32_16x16x32_f16(false, va0, false, vb0,
                                                 (short)0, acc, false, false);
    v16h va1 = frag16(arow, 4 + lh, 6 + lh);
    v16h vb1 = frag16(brow, 4 + lh, 6 + lh);
    acc = __builtin_amdgcn_wmma_f32_16x16x32_f16(false, va1, false, vb1,
                                                 (short)0, acc, false, false);
    __syncthreads();
    st ^= 1;
  }

  int col = n0 + lr;
  if (col < N) {
    float bv = bias ? bias[col] : 0.f;
    long long cb = (long long)col * c_ns;
    int row0 = mblk + wave * 16 + lh * 8;   // C layout: VGPR v -> M=row0+v
    if (c_ms == 1 && ((c_ns & 3) == 0) && row0 + 8 <= M) {
      float4 lo = { acc[0] * alpha + bv, acc[1] * alpha + bv,
                    acc[2] * alpha + bv, acc[3] * alpha + bv };
      float4 hi = { acc[4] * alpha + bv, acc[5] * alpha + bv,
                    acc[6] * alpha + bv, acc[7] * alpha + bv };
      *(float4*)&Cb[cb + row0]     = lo;
      *(float4*)&Cb[cb + row0 + 4] = hi;
    } else {
#pragma unroll
      for (int v = 0; v < 8; ++v) {
        int row = row0 + v;
        if (row < M)
          Cb[(long long)row * c_ms + cb] = acc[v] * alpha + bv;
      }
    }
  }
}

// ---------------- depthwise 4x4 conv (generic stride / pad-before) --------
__global__ void k_dwconv(const float* __restrict__ x, const float* __restrict__ w,
                         const float* __restrict__ b, float* __restrict__ y,
                         int Bn, int C, int Hi, int Wi, int Ho, int Wo,
                         int sh, int sw, int ph, int pw)
{
  long long idx = (long long)blockIdx.x * blockDim.x + threadIdx.x;
  long long tot = (long long)Bn * C * Ho * Wo;
  if (idx >= tot) return;
  int wo = (int)(idx % Wo); long long t = idx / Wo;
  int ho = (int)(t % Ho);   t /= Ho;
  int c  = (int)(t % C);    int bb = (int)(t / C);
  const float* xp = x + (long long)(bb * C + c) * Hi * Wi;
  const float* wp = w + c * 16;
  float acc = b[c];
#pragma unroll
  for (int kh = 0; kh < 4; ++kh) {
    int hi = ho * sh - ph + kh;
    if (hi < 0 || hi >= Hi) continue;
#pragma unroll
    for (int kw = 0; kw < 4; ++kw) {
      int wi = wo * sw - pw + kw;
      if (wi < 0 || wi >= Wi) continue;
      acc += xp[(long long)hi * Wi + wi] * wp[kh * 4 + kw];
    }
  }
  y[idx] = acc;
}

// ---------------- channel norm (over C at each (b,s)) ---------------------
__global__ void k_chnorm(const float* __restrict__ x, const float* __restrict__ g,
                         const float* __restrict__ be, float* __restrict__ y,
                         int C, long long S, int Bn)
{
  long long idx = (long long)blockIdx.x * blockDim.x + threadIdx.x;
  long long tot = (long long)Bn * S;
  if (idx >= tot) return;
  long long s = idx % S; int bb = (int)(idx / S);
  const float* xp = x + (long long)bb * C * S + s;
  float m = 0.f;
  for (int c = 0; c < 64; ++c) m += xp[(long long)c * S];
  m *= (1.f / 64.f);
  float v = 0.f;
  for (int c = 0; c < 64; ++c) { float d = xp[(long long)c * S] - m; v += d * d; }
  v *= (1.f / 64.f);
  float inv = rsqrtf(v + 1e-5f);
  float* yp = y + (long long)bb * C * S + s;
  for (int c = 0; c < 64; ++c)
    yp[(long long)c * S] = (xp[(long long)c * S] - m) * inv * g[c] + be[c];
}

// ---------------- group norm (whole sample): stats + apply ----------------
__global__ void k_zero(float* p, long long n)
{
  long long i = (long long)blockIdx.x * blockDim.x + threadIdx.x;
  if (i < n) p[i] = 0.f;
}

__global__ __launch_bounds__(256) void k_gnstats(const float* __restrict__ x,
                                                 float* stats, long long per)
{
  int bb = blockIdx.y;
  const float* p = x + (long long)bb * per;
  float s = 0.f, s2 = 0.f;
  for (long long i = (long long)blockIdx.x * 256 + threadIdx.x; i < per;
       i += (long long)gridDim.x * 256) {
    float v = p[i]; s += v; s2 += v * v;
  }
  __shared__ float r1[256], r2[256];
  int tid = threadIdx.x;
  r1[tid] = s; r2[tid] = s2; __syncthreads();
  for (int st = 128; st > 0; st >>= 1) {
    if (tid < st) { r1[tid] += r1[tid + st]; r2[tid] += r2[tid + st]; }
    __syncthreads();
  }
  if (tid == 0) { atomicAdd(&stats[bb * 2], r1[0]); atomicAdd(&stats[bb * 2 + 1], r2[0]); }
}

__global__ void k_gnapply(const float* __restrict__ x, const float* __restrict__ stats,
                          const float* __restrict__ g, const float* __restrict__ b,
                          float* __restrict__ y, int C, long long S, long long per, int Bn)
{
  long long idx = (long long)blockIdx.x * blockDim.x + threadIdx.x;
  long long tot = (long long)Bn * per;
  if (idx >= tot) return;
  int bb = (int)(idx / per); long long rem = idx % per;
  int c = (int)(rem / S);
  float mean = stats[bb * 2] / (float)per;
  float var  = stats[bb * 2 + 1] / (float)per - mean * mean;
  y[idx] = (x[idx] - mean) * rsqrtf(var + 1e-5f) * g[c] + b[c];
}

// ---------------- avg-pool(big)->(ho,wo) + base ---------------------------
__global__ void k_pool_add(const float* __restrict__ big, const float* __restrict__ base,
                           float* __restrict__ out, int C, int H, int W, int ho, int wo, int Bn)
{
  long long idx = (long long)blockIdx.x * blockDim.x + threadIdx.x;
  long long tot = (long long)Bn * C * ho * wo;
  if (idx >= tot) return;
  int j = (int)(idx % wo); long long q = idx / wo;
  int i = (int)(q % ho);   long long bc = q / ho;
  int fh = H / ho, fw = W / wo;
  const float* bp = big + bc * (long long)H * W;
  float s = 0.f;
  for (int a = 0; a < fh; ++a)
    for (int c2 = 0; c2 < fw; ++c2)
      s += bp[(long long)(i * fh + a) * W + (j * fw + c2)];
  out[idx] = base[idx] + s / (float)(fh * fw);
}

// ---------------- unfold: xn(B,C,T,F) -> u(L, B*T, C*8) -------------------
__global__ void k_unfold(const float* __restrict__ x, float* __restrict__ u,
                         int Bn, int C, int T_, int F_, int L)
{
  long long idx = (long long)blockIdx.x * blockDim.x + threadIdx.x;
  long long tot = (long long)L * Bn * T_ * C * 8;
  if (idx >= tot) return;
  int j = (int)(idx % 8); long long q = idx / 8;
  int c = (int)(q % C);   q /= C;
  long long n = q % ((long long)Bn * T_);
  int l = (int)(q / ((long long)Bn * T_));
  int bb = (int)(n / T_), tt = (int)(n % T_);
  u[idx] = x[(((long long)bb * C + c) * T_ + tt) * F_ + (j + l)];
}

// ---------------- SRU gates + scan ----------------------------------------
__global__ void k_sru_act(const float* __restrict__ U, const float* __restrict__ bv,
                          float* __restrict__ f, float* __restrict__ r, long long M)
{
  long long idx = (long long)blockIdx.x * blockDim.x + threadIdx.x;
  if (idx >= M * 32) return;
  int hh = (int)(idx % 32); long long m = idx / 32;
  f[idx] = sigmoidf_(U[m * 128 + 32 + hh] + bv[hh]);
  r[idx] = sigmoidf_(U[m * 128 + 64 + hh] + bv[32 + hh]);
}

// rev=1 scans l from L-1..0 (== reversed-input SRU with un-reversed output)
__global__ void k_sru_scan(const float* __restrict__ U, const float* __restrict__ f,
                           const float* __restrict__ r, float* __restrict__ h,
                           int L, int N, int rev, int off)
{
  long long idx = (long long)blockIdx.x * blockDim.x + threadIdx.x;
  if (idx >= (long long)N * 32) return;
  int hh = (int)(idx % 32); int n = (int)(idx / 32);
  float c = 0.f;
  for (int s = 0; s < L; ++s) {
    int l = rev ? (L - 1 - s) : s;
    long long row = (long long)l * N + n;
    float ff = f[row * 32 + hh];
    float rr = r[row * 32 + hh];
    float xt = U[row * 128 + hh];
    float xs = U[row * 128 + 96 + hh];
    c = ff * c + (1.f - ff) * xt;
    h[row * 64 + off + hh] = rr * c + (1.f - rr) * xs;
  }
}

// ---------------- ConvTranspose1d (stride1, pad0), h (L,N,64) -> y (N,64,Fo)
__global__ void k_convt(const float* __restrict__ h, const float* __restrict__ w,
                        const float* __restrict__ b, float* __restrict__ y,
                        int N, int L, int Fo)
{
  long long idx = (long long)blockIdx.x * blockDim.x + threadIdx.x;
  long long tot = (long long)N * 64 * Fo;
  if (idx >= tot) return;
  int t = (int)(idx % Fo); long long q = idx / Fo;
  int co = (int)(q % 64); int n = (int)(q / 64);
  float acc = b[co];
  for (int k = 0; k < 8; ++k) {
    int l = t - k;
    if (l < 0 || l >= L) continue;
    const float* hp = h + ((long long)l * N + n) * 64;
    for (int ci = 0; ci < 64; ++ci)
      acc += hp[ci] * w[(ci * 64 + co) * 8 + k];
  }
  y[idx] = acc;
}

// out(B,C,T,F) = y((b*T+t),c,f) + xn(B,C,T,F)
__global__ void k_dprnn_out(const float* __restrict__ y, const float* __restrict__ xn,
                            float* __restrict__ out, int C, int T_, int F_, int Bn)
{
  long long idx = (long long)blockIdx.x * blockDim.x + threadIdx.x;
  long long tot = (long long)Bn * C * T_ * F_;
  if (idx >= tot) return;
  int f = (int)(idx % F_); long long q = idx / F_;
  int t = (int)(q % T_);   q /= T_;
  int c = (int)(q % C);    int bb = (int)(q / C);
  out[idx] = y[(((long long)bb * T_ + t) * C + c) * F_ + f] + xn[idx];
}

// (B,C,H,W) -> (B,C,W,H)
__global__ void k_perm(const float* __restrict__ x, float* __restrict__ y,
                       int C, int H, int W, int Bn)
{
  long long idx = (long long)blockIdx.x * blockDim.x + threadIdx.x;
  long long tot = (long long)Bn * C * H * W;
  if (idx >= tot) return;
  int w = (int)(idx % W); long long q = idx / W;
  int h = (int)(q % H);   long long bc = q / H;
  y[(bc * W + w) * (long long)H + h] = x[idx];
}

__global__ void k_addpe(const float* __restrict__ x, const float* __restrict__ pe,
                        float* __restrict__ y, int C, int T_, int F_, int Bn)
{
  long long idx = (long long)blockIdx.x * blockDim.x + threadIdx.x;
  long long tot = (long long)Bn * C * T_ * F_;
  if (idx >= tot) return;
  int f = (int)(idx % F_);
  int c = (int)((idx / ((long long)F_ * T_)) % C);
  y[idx] = x[idx] + pe[c * F_ + f];
}

__global__ __launch_bounds__(256) void k_softmax(float* __restrict__ x, int ncol)
{
  long long row = blockIdx.x;
  float* p = x + row * (long long)ncol;
  __shared__ float red[256];
  int tid = threadIdx.x;
  float m = -1e30f;
  for (int j = tid; j < ncol; j += 256) m = fmaxf(m, p[j]);
  red[tid] = m; __syncthreads();
  for (int s = 128; s > 0; s >>= 1) {
    if (tid < s) red[tid] = fmaxf(red[tid], red[tid + s]);
    __syncthreads();
  }
  m = red[0]; __syncthreads();
  float sum = 0.f;
  for (int j = tid; j < ncol; j += 256) { float e = __expf(p[j] - m); p[j] = e; sum += e; }
  red[tid] = sum; __syncthreads();
  for (int s = 128; s > 0; s >>= 1) {
    if (tid < s) red[tid] += red[tid + s];
    __syncthreads();
  }
  float inv = 1.f / red[0];
  for (int j = tid; j < ncol; j += 256) p[j] *= inv;
}

// out = sigmoid(interp(w1)) * w2 + interp(w3)
__global__ void k_tfar_comb(const float* __restrict__ w1, const float* __restrict__ w2,
                            const float* __restrict__ w3, float* __restrict__ out,
                            int C, int Hm, int Wm, int Hn, int Wn, int Bn)
{
  long long idx = (long long)blockIdx.x * blockDim.x + threadIdx.x;
  long long tot = (long long)Bn * C * Hm * Wm;
  if (idx >= tot) return;
  int w = (int)(idx % Wm); long long q = idx / Wm;
  int h = (int)(q % Hm);   long long bc = q / Hm;
  int hi = (int)(((long long)h * Hn) / Hm);
  int wi = (int)(((long long)w * Wn) / Wm);
  long long nidx = (bc * Hn + hi) * (long long)Wn + wi;
  out[idx] = sigmoidf_(w1[nidx]) * w2[idx] + w3[nidx];
}

__global__ void k_add(const float* __restrict__ a, const float* __restrict__ b,
                      float* __restrict__ y, long long n)
{
  long long i = (long long)blockIdx.x * blockDim.x + threadIdx.x;
  if (i < n) y[i] = a[i] + b[i];
}

__global__ void k_addinto(float* __restrict__ y, const float* __restrict__ x, long long n)
{
  long long i = (long long)blockIdx.x * blockDim.x + threadIdx.x;
  if (i < n) y[i] += x[i];
}

// ======================================================================
// Host side
// ======================================================================
static inline dim3 g1(long long n) { return dim3((unsigned)((n + 255) / 256)); }

static void launch_gemm(hipStream_t s, const float* A, const float* B, float* C,
                        const float* bias, int M, int N, int K, float alpha,
                        long long ams, long long aks, long long abo, long long abi,
                        long long bks, long long bns, long long bbo, long long bbi,
                        long long cms, long long cns, long long cbo, long long cbi,
                        int nbo, int nbi)
{
  dim3 grid((N + 15) / 16, (M + 63) / 64, nbo * nbi);
  k_gemm_wmma<<<grid, dim3(128), 0, s>>>(A, B, C, bias, M, N, K, alpha,
      ams, aks, abo, abi, bks, bns, bbo, bbi, cms, cns, cbo, cbi, nbi);
}

// Input flattening: setup_inputs() dict insertion order, recursed.
enum {
  IN_AUDIO = 0, IN_COMP_W, IN_COMP_B,
  IN_PH0_W, IN_PH0_B, IN_PH1_W, IN_PH1_B,
  IN_FD = 7,                       // g,beta,(Wf,bf,Wb,bb)x4,ct_w,ct_b
  IN_TD = 27,
  IN_QW = 47, IN_QB, IN_KW, IN_KB, IN_VW, IN_VB, IN_OW, IN_OB, IN_PE,
  IN_FWD0 = 56, IN_FWD1 = 68, IN_BWD0 = 80,
  IN_OUT_W = 92, IN_OUT_B = 93
};

extern "C" void kernel_launch(void* const* d_in, const int* in_sizes, int n_in,
                              void* d_out, int out_size, void* d_ws, size_t ws_size,
                              hipStream_t stream)
{
  (void)in_sizes; (void)n_in; (void)out_size; (void)ws_size;
  auto P = [&](int i) { return (const float*)d_in[i]; };

  float* ws = (float*)d_ws;
  size_t off = 0;
  auto alloc = [&](size_t n) { float* p = ws + off; off += (n + 255) & ~(size_t)255; return p; };

  const float* audio = P(IN_AUDIO);
  const long long S0 = 65536;   // 512*128
  const long long Sa = 16384;   // 256*64

  float* comp  = alloc(8388608);   // (2,64,512,128)
  float* c0    = alloc(8388608);   // compressed[0]
  float* c1    = alloc(2097152);   // compressed[1] (2,64,256,64)
  float* cs    = alloc(2097152);   // compress after pool-add
  float* xn    = alloc(2097152);   // dprnn channel-norm (shared freq/time)
  float* ub    = alloc(16318464);  // unfold buffer (max of both dprnns)
  float* Ub    = alloc(4079616);   // SRU projection (maxM * 128)
  float* fb    = alloc(1019904);
  float* rb    = alloc(1019904);
  float* hA    = alloc(2039808);
  float* hB    = alloc(2039808);
  float* yb    = alloc(2097152);   // convT output
  float* fqout = alloc(2097152);
  float* fpt   = alloc(2097152);
  float* tdout = alloc(2097152);
  float* tp    = alloc(2097152);
  float* xin   = alloc(2097152);
  float* qb_   = alloc(2097152);
  float* kb_   = alloc(2097152);
  float* vb_   = alloc(2097152);
  float* sc    = alloc(8388608);   // scores (2,64,256,256)
  float* ao    = alloc(2097152);
  float* pj    = alloc(2097152);
  float* att   = alloc(2097152);
  float* w1b   = alloc(2097152);
  float* w2b   = alloc(8388608);
  float* w3b   = alloc(2097152);
  float* tmpb  = alloc(8388608);
  float* stats = alloc(256);
  float* ups0  = alloc(8388608);
  float* ups1  = alloc(2097152);
  float* upf   = alloc(8388608);

  // ---------------- DPRNN -------------------------------------------------
  auto run_dprnn = [&](const float* xsrc, int T_, int F_, int pbase, float* outb) {
    int Lh = F_ - 7;
    int N  = 2 * T_;
    int Mh = Lh * N;
    long long S_ = (long long)T_ * F_;
    k_chnorm<<<g1(2LL * S_), 256, 0, stream>>>(xsrc, P(pbase + 0), P(pbase + 1), xn, 64, S_, 2);
    k_unfold<<<g1((long long)Mh * 512), 256, 0, stream>>>(xn, ub, 2, 64, T_, F_, Lh);
    const float* x = ub; long long xms = 512; int din = 512;
    float* hcur = hA;
    for (int layer = 0; layer < 4; ++layer) {
      int pb = pbase + 2 + layer * 4;   // Wf, bf, Wb, bb
      for (int dir = 0; dir < 2; ++dir) {
        launch_gemm(stream, x, P(pb + (dir ? 2 : 0)), Ub, nullptr, Mh, 128, din, 1.f,
                    xms, 1, 0, 0,   128, 1, 0, 0,   128, 1, 0, 0, 1, 1);
        k_sru_act<<<g1((long long)Mh * 32), 256, 0, stream>>>(Ub, P(pb + (dir ? 3 : 1)),
                                                              fb, rb, (long long)Mh);
        k_sru_scan<<<g1((long long)N * 32), 256, 0, stream>>>(Ub, fb, rb, hcur, Lh, N,
                                                              dir, dir ? 32 : 0);
      }
      x = hcur; xms = 64; din = 64;
      hcur = (x == hA) ? hB : hA;
    }
    k_convt<<<g1((long long)N * 64 * F_), 256, 0, stream>>>(x, P(pbase + 18), P(pbase + 19),
                                                            yb, N, Lh, F_);
    k_dprnn_out<<<g1(2LL * 64 * S_), 256, 0, stream>>>(yb, xn, outb, 64, T_, F_, 2);
  };

  // ---------------- TFAR helpers ------------------------------------------
  auto branch = [&](const float* xsrc, int H, int W, int pb, float* dst) {
    long long tot = 2LL * 64 * H * W;
    k_dwconv<<<g1(tot), 256, 0, stream>>>(xsrc, P(pb), P(pb + 1), tmpb,
                                          2, 64, H, W, H, W, 1, 1, 1, 1);
    k_zero<<<dim3(1), dim3(64), 0, stream>>>(stats, 8);
    long long per = 64LL * H * W;
    long long nblk = (per + 255) / 256; if (nblk > 1024) nblk = 1024;
    dim3 gs((unsigned)nblk, 2);
    k_gnstats<<<gs, 256, 0, stream>>>(tmpb, stats, per);
    k_gnapply<<<g1(tot), 256, 0, stream>>>(tmpb, stats, P(pb + 2), P(pb + 3), dst,
                                           64, (long long)H * W, per, 2);
  };
  auto tfar = [&](const float* m, int Hm, int Wm, const float* n, int Hn, int Wn,
                  int pbase, float* outb) {
    branch(n, Hn, Wn, pbase + 0, w1b);
    branch(m, Hm, Wm, pbase + 4, w2b);
    branch(n, Hn, Wn, pbase + 8, w3b);
    k_tfar_comb<<<g1(2LL * 64 * Hm * Wm), 256, 0, stream>>>(w1b, w2b, w3b, outb,
                                                            64, Hm, Wm, Hn, Wn, 2);
  };

  // ===================== pipeline ==========================================
  // 1) compress = conv1x1(audio)  (GEMM: M=spatial, K=256, N=64, batched over B)
  launch_gemm(stream, audio, P(IN_COMP_W), comp, P(IN_COMP_B), 65536, 64, 256, 1.f,
              1, S0, 256 * S0, 0,   1, 256, 0, 0,   1, S0, 64 * S0, 0, 2, 1);
  // 2) phase convs (depthwise) + avg-pool skip sum
  k_dwconv<<<g1(8388608LL), 256, 0, stream>>>(comp, P(IN_PH0_W), P(IN_PH0_B), c0,
                                              2, 64, 512, 128, 512, 128, 1, 1, 1, 1);
  k_dwconv<<<g1(2097152LL), 256, 0, stream>>>(c0, P(IN_PH1_W), P(IN_PH1_B), c1,
                                              2, 64, 512, 128, 256, 64, 2, 2, 1, 1);
  k_pool_add<<<g1(2097152LL), 256, 0, stream>>>(c0, c1, cs, 64, 512, 128, 256, 64, 2);
  // 3) freq DPRNN, transpose, time DPRNN, transpose back
  run_dprnn(cs, 256, 64, IN_FD, fqout);
  k_perm<<<g1(2097152LL), 256, 0, stream>>>(fqout, fpt, 64, 256, 64, 2);
  run_dprnn(fpt, 64, 256, IN_TD, tdout);
  k_perm<<<g1(2097152LL), 256, 0, stream>>>(tdout, tp, 64, 64, 256, 2);
  // 4) TF attention (per-frequency over time), residual
  k_addpe<<<g1(2097152LL), 256, 0, stream>>>(tp, P(IN_PE), xin, 64, 256, 64, 2);
  launch_gemm(stream, xin, P(IN_QW), qb_, P(IN_QB), 16384, 64, 64, 1.f,
              1, Sa, 64 * Sa, 0,   1, 64, 0, 0,   1, Sa, 64 * Sa, 0, 2, 1);
  launch_gemm(stream, xin, P(IN_KW), kb_, P(IN_KB), 16384, 64, 64, 1.f,
              1, Sa, 64 * Sa, 0,   1, 64, 0, 0,   1, Sa, 64 * Sa, 0, 2, 1);
  launch_gemm(stream, xin, P(IN_VW), vb_, P(IN_VB), 16384, 64, 64, 1.f,
              1, Sa, 64 * Sa, 0,   1, 64, 0, 0,   1, Sa, 64 * Sa, 0, 2, 1);
  // scores[b,f,t,s] = (1/8) * sum_c q[b,c,t,f] k[b,c,s,f]   (batch = (b,f))
  launch_gemm(stream, qb_, kb_, sc, nullptr, 256, 256, 64, 0.125f,
              64, Sa, 64 * Sa, 1,   Sa, 64, 64 * Sa, 1,
              256, 1, 64LL * 256 * 256, 256LL * 256, 2, 64);
  k_softmax<<<dim3(2 * 64 * 256), 256, 0, stream>>>(sc, 256);
  // out[b,c,t,f] = sum_s attn[b,f,t,s] v[b,c,s,f]
  launch_gemm(stream, sc, vb_, ao, nullptr, 256, 64, 256, 1.f,
              256, 1, 64LL * 256 * 256, 256LL * 256,
              64, Sa, 64 * Sa, 1,   64, Sa, 64 * Sa, 1, 2, 64);
  launch_gemm(stream, ao, P(IN_OW), pj, P(IN_OB), 16384, 64, 64, 1.f,
              1, Sa, 64 * Sa, 0,   1, 64, 0, 0,   1, Sa, 64 * Sa, 0, 2, 1);
  k_add<<<g1(2097152LL), 256, 0, stream>>>(tp, pj, att, 2097152LL);
  // 5) TFAR fusion path
  tfar(c0, 512, 128, att, 256, 64, IN_FWD0, ups0);
  tfar(c1, 256, 64,  att, 256, 64, IN_FWD1, ups1);
  tfar(ups0, 512, 128, ups1, 256, 64, IN_BWD0, upf);
  k_addinto<<<g1(8388608LL), 256, 0, stream>>>(upf, c0, 8388608LL);
  // 6) out conv + audio residual
  float* outp = (float*)d_out;
  launch_gemm(stream, upf, P(IN_OUT_W), outp, P(IN_OUT_B), 65536, 256, 64, 1.f,
              1, S0, 64 * S0, 0,   1, 64, 0, 0,   1, S0, 256 * S0, 0, 2, 1);
  k_addinto<<<g1(33554432LL), 256, 0, stream>>>(outp, audio, 33554432LL);
}
